// GraphEncoder_86311662780919
// MI455X (gfx1250) — matrix operand order repaired and checked by reading, not compiled, not run on hardware
//
#include <hip/hip_runtime.h>
#include <hip/hip_bf16.h>
#include <stdint.h>
#include <stddef.h>

#define N 8192
#define KS 4            // K-split factor for the big SpMM
#define NT (N / 16)     // 512 row tiles

typedef __attribute__((ext_vector_type(16))) __bf16 v16bf;
typedef __attribute__((ext_vector_type(8)))  float  v8f;

union V16 {
    v16bf v;
    __bf16 h[16];
    unsigned short u[16];
    uint4 q[2];
};
union Acc {
    v8f v;
    float f[8];
};

__device__ __forceinline__ unsigned short f2bf(float f) {
    union { __bf16 h; unsigned short u; } cv;
    cv.h = (__bf16)f;                      // RNE via native cvt
    return cv.u;
}

__device__ __forceinline__ void pack4h(__bf16* h, float4 f) {
    h[0] = (__bf16)f.x; h[1] = (__bf16)f.y; h[2] = (__bf16)f.z; h[3] = (__bf16)f.w;
}

__device__ __forceinline__ v16bf loadB16(const unsigned short* p) {
    V16 b;
    const uint4* q = (const uint4*)p;
    b.q[0] = q[0]; b.q[1] = q[1];
    return b.v;
}

// ---------------- Kernel 1: deg[i] = sum_j adj[i,j]  (adj pass 1) -------------
__global__ void k_rowsum(const float* __restrict__ adj, float* __restrict__ deg) {
    __shared__ float red[256];
    const int row = blockIdx.x;
    const float4* p4 = (const float4*)(adj + (size_t)row * N);
    float s = 0.f;
    for (int j = threadIdx.x; j < N / 4; j += 256) {
        float4 v = p4[j];
        s += (v.x + v.y) + (v.z + v.w);
    }
    red[threadIdx.x] = s;
    __syncthreads();
    for (int o = 128; o > 0; o >>= 1) {
        if (threadIdx.x < o) red[threadIdx.x] += red[threadIdx.x + o];
        __syncthreads();
    }
    if (threadIdx.x == 0) deg[row] = red[0];
}

// ---------------- Kernel 2: dinv, dinv*deg_norm, W2^T bf16 --------------------
__global__ void k_prep(const float* __restrict__ deg, const float* __restrict__ W2,
                       float* __restrict__ dinv, float* __restrict__ w1w,
                       unsigned short* __restrict__ W2T) {
    __shared__ float red[1024];
    float m = 0.f;
    for (int i = threadIdx.x; i < N; i += 1024) m = fmaxf(m, deg[i]);
    red[threadIdx.x] = m;
    __syncthreads();
    for (int o = 512; o > 0; o >>= 1) {
        if (threadIdx.x < o) red[threadIdx.x] = fmaxf(red[threadIdx.x], red[threadIdx.x + o]);
        __syncthreads();
    }
    const float M = fmaxf(red[0], 1.0f);
    for (int i = threadIdx.x; i < N; i += 1024) {
        float d  = deg[i];
        float di = rsqrtf(fmaxf(d + 1.0f, 1.0f));
        dinv[i] = di;
        w1w[i]  = di * (d / M);
    }
    // W2T[n][k] = bf16(W2[k][n])  (B-operand wants contiguous K per column)
    for (int i = threadIdx.x; i < 64 * 64; i += 1024) {
        int n = i >> 6, k = i & 63;
        W2T[n * 64 + k] = f2bf(W2[k * 64 + n]);
    }
}

// ---------------- Kernel 3: layer 1 fused  (adj pass 2) -----------------------
// s0 = sum_j adj[i,j]*dinv[j], s1 = sum_j adj[i,j]*w1w[j]
// y_c = dinv[i]*(s_c + w_c[i]);  h1 = relu(y@W1 + b1);  gT = (dinv[i]*h1)^T bf16
__global__ void k_layer1(const float* __restrict__ adj, const float* __restrict__ dinv,
                         const float* __restrict__ w1w, const float* __restrict__ W1,
                         const float* __restrict__ b1, float* __restrict__ h1f,
                         unsigned short* __restrict__ gT) {
    __shared__ float r0[256], r1[256];
    const int row = blockIdx.x;
    const float4* p4 = (const float4*)(adj + (size_t)row * N);
    const float4* d4 = (const float4*)dinv;
    const float4* w4 = (const float4*)w1w;
    float s0 = 0.f, s1 = 0.f;
    for (int j = threadIdx.x; j < N / 4; j += 256) {
        float4 a = p4[j];
        float4 d = d4[j];
        float4 w = w4[j];
        s0 += a.x * d.x + a.y * d.y + a.z * d.z + a.w * d.w;
        s1 += a.x * w.x + a.y * w.y + a.z * w.z + a.w * w.w;
    }
    r0[threadIdx.x] = s0; r1[threadIdx.x] = s1;
    __syncthreads();
    for (int o = 128; o > 0; o >>= 1) {
        if (threadIdx.x < o) {
            r0[threadIdx.x] += r0[threadIdx.x + o];
            r1[threadIdx.x] += r1[threadIdx.x + o];
        }
        __syncthreads();
    }
    if (threadIdx.x < 64) {
        float di = dinv[row];
        float y0 = di * (r0[0] + di);          // w0[row] == dinv[row]
        float y1 = di * (r1[0] + w1w[row]);
        int c = threadIdx.x;
        float h = fmaf(y0, W1[c], fmaf(y1, W1[64 + c], b1[c]));
        h = fmaxf(h, 0.f);
        h1f[(size_t)row * 64 + c] = h;
        gT[(size_t)c * N + row] = f2bf(di * h); // g = D*h1, stored transposed
    }
}

// ---------------- Kernel 4: zpart = adj @ g  via WMMA  (adj pass 3) -----------
// One wave computes a 16(M) x 64(N) tile over one K-split segment.
// Software-pipelined: iteration k+1 loads are issued before iteration k WMMAs.
__global__ void k_spmm(const float* __restrict__ adj, const unsigned short* __restrict__ gT,
                       float* __restrict__ zpart) {
    const int tile = blockIdx.x;          // 0..NT-1
    const int ks   = blockIdx.y;          // 0..KS-1
    const int lane = threadIdx.x;         // 0..31
    const int half = lane >> 4;           // 0 / 1
    const int l16  = lane & 15;
    const int row0 = tile * 16;
    const int akb  = half * 16;           // A K-base within 32 for this half
    const int bkb  = half * 16;           // B K-base within 32 for this half

    Acc c0, c1, c2, c3;
    #pragma unroll
    for (int r = 0; r < 8; ++r) { c0.f[r] = 0.f; c1.f[r] = 0.f; c2.f[r] = 0.f; c3.f[r] = 0.f; }

    const float* arow = adj + (size_t)(row0 + l16) * N;  // A: lane = row M
    const unsigned short* g0r = gT + (size_t)( 0 + l16) * N;
    const unsigned short* g1r = gT + (size_t)(16 + l16) * N;
    const unsigned short* g2r = gT + (size_t)(32 + l16) * N;
    const unsigned short* g3r = gT + (size_t)(48 + l16) * N;

    const int k_begin = ks * (N / KS);
    const int k_end   = k_begin + (N / KS);

    // ---- preload first iteration ----
    float4 f0, f1, f2, f3;
    V16 b0, b1, b2, b3;
    {
        const float4* pa = (const float4*)(arow + k_begin + akb);
        f0 = pa[0]; f1 = pa[1];
        const float4* pb = (const float4*)(arow + k_begin + akb + 8);
        f2 = pb[0]; f3 = pb[1];
        const size_t bko = (size_t)(k_begin + bkb);
        b0.v = loadB16(g0r + bko);
        b1.v = loadB16(g1r + bko);
        b2.v = loadB16(g2r + bko);
        b3.v = loadB16(g3r + bko);
    }

    for (int k0 = k_begin; k0 < k_end; k0 += 32) {
        // ---- issue next-iteration loads (wraps once at the end; harmless) ----
        const int kn = (k0 + 32 < k_end) ? (k0 + 32) : k_begin;
        float4 nf0, nf1, nf2, nf3;
        V16 nb0, nb1, nb2, nb3;
        {
            const float4* pa = (const float4*)(arow + kn + akb);
            nf0 = pa[0]; nf1 = pa[1];
            const float4* pb = (const float4*)(arow + kn + akb + 8);
            nf2 = pb[0]; nf3 = pb[1];
            const size_t bko = (size_t)(kn + bkb);
            nb0.v = loadB16(g0r + bko);
            nb1.v = loadB16(g1r + bko);
            nb2.v = loadB16(g2r + bko);
            nb3.v = loadB16(g3r + bko);
        }
        // pull the HBM adj stream further ahead (global_prefetch, counter-free)
        __builtin_prefetch(arow + k0 + 256, 0, 1);

        // ---- convert current A tile (packed f32->bf16 cvt, RNE) ----
        V16 a;
        pack4h(&a.h[0],  f0); pack4h(&a.h[4],  f1);
        pack4h(&a.h[8],  f2); pack4h(&a.h[12], f3);

        c0.v = __builtin_amdgcn_wmma_f32_16x16x32_bf16(false, a.v, false, b0.v, (short)0, c0.v, false, false);
        c1.v = __builtin_amdgcn_wmma_f32_16x16x32_bf16(false, a.v, false, b1.v, (short)0, c1.v, false, false);
        c2.v = __builtin_amdgcn_wmma_f32_16x16x32_bf16(false, a.v, false, b2.v, (short)0, c2.v, false, false);
        c3.v = __builtin_amdgcn_wmma_f32_16x16x32_bf16(false, a.v, false, b3.v, (short)0, c3.v, false, false);

        f0 = nf0; f1 = nf1; f2 = nf2; f3 = nf3;
        b0 = nb0; b1 = nb1; b2 = nb2; b3 = nb3;
    }

    // ---- store partial tile in C layout order: zpart[ks][tile][m][n] ----
    float* zp = zpart + ((size_t)(ks * NT + tile) * 16) * 64;
    #pragma unroll
    for (int r = 0; r < 8; ++r) {
        int mm = r + half * 8;
        zp[mm * 64 +  0 + l16] = c0.f[r];
        zp[mm * 64 + 16 + l16] = c1.f[r];
        zp[mm * 64 + 32 + l16] = c2.f[r];
        zp[mm * 64 + 48 + l16] = c3.f[r];
    }
}

// ---------------- Kernel 5: z = D*t + D^2*h1;  out = z @ W2 + b2 (WMMA) -------
__global__ void k_epilogue(const float* __restrict__ zpart, const float* __restrict__ dinv,
                           const float* __restrict__ h1f, const unsigned short* __restrict__ W2T,
                           const float* __restrict__ b2, float* __restrict__ out) {
    __shared__ unsigned short zb[16 * 64];
    const int tile = blockIdx.x;
    const int lane = threadIdx.x;
    const int half = lane >> 4;
    const int l16  = lane & 15;
    const int row0 = tile * 16;
    const int akb  = half * 16;
    const int bkb  = half * 16;

    float dl[8];
    #pragma unroll
    for (int r = 0; r < 8; ++r) dl[r] = dinv[row0 + r + half * 8];

    // reduce K-split partials, scale, add diagonal term, park as bf16 in LDS
    #pragma unroll
    for (int nb = 0; nb < 4; ++nb) {
        int n = nb * 16 + l16;
        #pragma unroll
        for (int r = 0; r < 8; ++r) {
            int mm = r + half * 8;
            float acc = 0.f;
            #pragma unroll
            for (int ks = 0; ks < KS; ++ks)
                acc += zpart[((size_t)(ks * NT + tile) * 16 + mm) * 64 + n];
            float di = dl[r];
            float z  = di * acc + di * di * h1f[(size_t)(row0 + mm) * 64 + n];
            zb[mm * 64 + n] = f2bf(z);
        }
    }
    __syncthreads();

    // out tile = z(16x64) @ W2(64x64) + b2 : 2 K-steps x 4 N-tiles of WMMA
    Acc acc[4];
    #pragma unroll
    for (int nb = 0; nb < 4; ++nb) {
        float bv = b2[nb * 16 + l16];
        #pragma unroll
        for (int r = 0; r < 8; ++r) acc[nb].f[r] = bv;   // C init = b2 broadcast
    }

    #pragma unroll
    for (int kb = 0; kb < 64; kb += 32) {
        V16 a;                                           // A from LDS, lane = row M = l16
        const uint4* pl = (const uint4*)&zb[l16 * 64 + kb + akb];
        a.q[0] = pl[0];                                  // K = akb..akb+7
        a.q[1] = *(const uint4*)&zb[l16 * 64 + kb + akb + 8];   // K = akb+8..+15
        #pragma unroll
        for (int nb = 0; nb < 4; ++nb) {
            int n = nb * 16 + l16;
            v16bf b = loadB16(W2T + (size_t)n * 64 + kb + bkb);
            acc[nb].v = __builtin_amdgcn_wmma_f32_16x16x32_bf16(false, a.v, false, b, (short)0, acc[nb].v, false, false);
        }
    }

    #pragma unroll
    for (int nb = 0; nb < 4; ++nb) {
        int n = nb * 16 + l16;
        #pragma unroll
        for (int r = 0; r < 8; ++r) {
            int mm = r + half * 8;
            out[(size_t)(row0 + mm) * 64 + n] = acc[nb].f[r];
        }
    }
}

// -----------------------------------------------------------------------------
extern "C" void kernel_launch(void* const* d_in, const int* in_sizes, int n_in,
                              void* d_out, int out_size, void* d_ws, size_t ws_size,
                              hipStream_t stream) {
    (void)in_sizes; (void)n_in; (void)out_size; (void)ws_size;
    const float* adj = (const float*)d_in[0];
    const float* W1  = (const float*)d_in[1];
    const float* b1  = (const float*)d_in[2];
    const float* W2  = (const float*)d_in[3];
    const float* b2  = (const float*)d_in[4];
    float* out = (float*)d_out;

    char* ws = (char*)d_ws;
    float*          deg   = (float*)(ws + 0);                       //  32 KB
    float*          dinv  = (float*)(ws + (32 << 10));              //  32 KB
    float*          w1w   = (float*)(ws + (64 << 10));              //  32 KB
    unsigned short* W2T   = (unsigned short*)(ws + (96 << 10));     //   8 KB
    unsigned short* gT    = (unsigned short*)(ws + (128 << 10));    //   1 MB
    float*          h1f   = (float*)(ws + (128 << 10) + (1 << 20)); //   2 MB
    float*          zpart = (float*)(ws + (128 << 10) + (3 << 20)); //   8 MB

    k_rowsum  <<<N, 256, 0, stream>>>(adj, deg);
    k_prep    <<<1, 1024, 0, stream>>>(deg, W2, dinv, w1w, W2T);
    k_layer1  <<<N, 256, 0, stream>>>(adj, dinv, w1w, W1, b1, h1f, gT);
    k_spmm    <<<dim3(NT, KS), 32, 0, stream>>>(adj, gT, zpart);
    k_epilogue<<<NT, 32, 0, stream>>>(zpart, dinv, h1f, W2T, b2, out);
}
